// Unfolded_RNN_54520314856026
// MI455X (gfx1250) — compile-verified, matrix operand order checked by reading, not run on hardware
//
#include <hip/hip_runtime.h>

// ---------------------------------------------------------------------------
// Unfolded LISTA-style RNN for MI455X (gfx1250), bf16 WMMA + f32 accumulate.
//
// Roofline: ~240 G MACs total, ~200MB working set (fits in 192MB L2,
// trivial vs 23.3 TB/s HBM) => compute bound => v_wmma_f32_16x16x32_bf16.
// Rewrites: pre_input is dead; xv_t = x_t@V^T hoisted out of the time loop;
// temp = D^T(A^T A)D is symmetric. All GEMMs in NT form C = X[M,K]*W[N,K]^T.
//
// Data movement: global->LDS staging uses gfx1250 async direct-to-LDS copies
// (GLOBAL_LOAD_ASYNC_TO_LDS_B128 + s_wait_asynccnt) with double-buffered LDS
// tiles so the DMA of tile k+1 overlaps the WMMAs of tile k. Builtin signature
// (from compiler diagnostics): (int4 AS1*, int4 AS3*, int offset, int cpol).
// ---------------------------------------------------------------------------

typedef unsigned short u16;
typedef __attribute__((ext_vector_type(16))) __bf16 bf16x16;
typedef __attribute__((ext_vector_type(8)))  float  floatx8;
typedef __attribute__((ext_vector_type(4)))  int    v4i;

#if defined(__AMDGCN__) && __has_builtin(__builtin_amdgcn_global_load_async_to_lds_b128) && \
    __has_builtin(__builtin_amdgcn_s_wait_asynccnt)
#define USE_ASYNC_LDS 1
typedef __attribute__((address_space(1))) v4i* gv4i_p;  // global int4*
typedef __attribute__((address_space(3))) v4i* lv4i_p;  // LDS int4*
#else
#define USE_ASYNC_LDS 0
#endif

__device__ __forceinline__ u16 f32_to_bf16(float f) {
  union { float f; unsigned u; } x; x.f = f;
  unsigned r = x.u + 0x7FFFu + ((x.u >> 16) & 1u);  // round-to-nearest-even
  return (u16)(r >> 16);
}

// multi-branch proximal operator of w0*|x| + w1*|x - a1|
__device__ __forceinline__ float prox_l1l1(float z, float w0, float w1, float a1) {
  float a0s, a1s, w0s, w1s;
  if (a1 >= 0.0f) { a0s = 0.0f; a1s = a1;   w0s = w0; w1s = w1; }
  else            { a0s = a1;   a1s = 0.0f; w0s = w1; w1s = w0; }
  if (z >= a1s + w0s + w1s) return z - w0s - w1s;
  if (z >= a1s + w0s - w1s) return a1s;
  if (z >= a0s + w0s - w1s) return z - w0s + w1s;
  if (z >= a0s - w0s - w1s) return a0s;
  return z + w0s + w1s;
}

// ---------------------------------------------------------------------------
// NT GEMM: C[M,N] = scale * (X[M,K] @ W[N,K]^T)  (bf16 operands, f32 accum)
// Block: 128x128x32, 256 threads = 8 waves (2 M-halves x 4 N-quarters),
// each wave: 4x2 grid of 16x16 WMMA accumulators.
// EPI: 0 = f32 store, 1 = bf16 store, 2 = fused prox -> bf16 store.
// ---------------------------------------------------------------------------
#define BM 128
#define BN 128
#define BK 32
#define LDSK 40  // padded row stride (elements) for LDS tiles

template <int EPI>
__global__ __launch_bounds__(256) void gemm_nt(
    const u16* __restrict__ X, const u16* __restrict__ W, void* __restrict__ outp,
    int M, int N, int K,
    const float* __restrict__ alpha_ptr, int inv_alpha,
    const float* __restrict__ xv, const float* __restrict__ a1,
    const float* __restrict__ lam0, const float* __restrict__ lam1) {
  const int tid  = threadIdx.x;
  const int lane = tid & 31;
  const int wid  = tid >> 5;
  const int wy = wid >> 2;  // 0..1 : which 64-row half
  const int wx = wid & 3;   // 0..3 : which 32-col quarter
  const int m0 = blockIdx.y * BM;
  const int n0 = blockIdx.x * BN;

  floatx8 acc[4][2];
  const floatx8 zero = {0.f, 0.f, 0.f, 0.f, 0.f, 0.f, 0.f, 0.f};
#pragma unroll
  for (int i = 0; i < 4; ++i) { acc[i][0] = zero; acc[i][1] = zero; }

  const int ml = lane & 15;  // row/col within 16-wide tile
  const int kh = lane >> 4;  // K-half select

  // Per-thread staging chunk coordinates: 4 x 16B chunks per thread per tile
  // (chunk c: row = c>>2 in [0,128), kcol = (c&3)*8 in {0,8,16,24}).
  const int rA0 = tid >> 2,         kA0 = (tid & 3) << 3;
  const int rA1 = (tid + 256) >> 2, kA1 = ((tid + 256) & 3) << 3;

  // One K-step of fragment loads + 8 WMMAs from the given LDS tiles.
  auto mma_step = [&](const u16* xs, const u16* ws) {
    union Frag { bf16x16 v; uint4 u[2]; };
    Frag fa[4], fb[2];
#pragma unroll
    for (int mt = 0; mt < 4; ++mt) {
      const u16* p = &xs[(wy * 64 + mt * 16 + ml) * LDSK + kh * 8];
      fa[mt].u[0] = *reinterpret_cast<const uint4*>(p);
      fa[mt].u[1] = *reinterpret_cast<const uint4*>(p + 16);
    }
#pragma unroll
    for (int nt = 0; nt < 2; ++nt) {
      const u16* p = &ws[(wx * 32 + nt * 16 + ml) * LDSK + kh * 8];
      fb[nt].u[0] = *reinterpret_cast<const uint4*>(p);
      fb[nt].u[1] = *reinterpret_cast<const uint4*>(p + 16);
    }
#pragma unroll
    for (int mt = 0; mt < 4; ++mt)
#pragma unroll
      for (int nt = 0; nt < 2; ++nt)
        acc[mt][nt] = __builtin_amdgcn_wmma_f32_16x16x32_bf16(
            false, fa[mt].v, false, fb[nt].v, (short)0, acc[mt][nt], false, false);
  };

#if USE_ASYNC_LDS
  // ---- async direct-to-LDS staging with double-buffered LDS tiles ----
  __shared__ __align__(16) u16 Xs[2][BM * LDSK];
  __shared__ __align__(16) u16 Ws[2][BN * LDSK];

  auto issue_tile = [&](int buf, int k0) {
    __builtin_amdgcn_global_load_async_to_lds_b128(
        (gv4i_p)(X + (size_t)(m0 + rA0) * K + k0 + kA0),
        (lv4i_p)(&Xs[buf][rA0 * LDSK + kA0]), 0, 0);
    __builtin_amdgcn_global_load_async_to_lds_b128(
        (gv4i_p)(X + (size_t)(m0 + rA1) * K + k0 + kA1),
        (lv4i_p)(&Xs[buf][rA1 * LDSK + kA1]), 0, 0);
    __builtin_amdgcn_global_load_async_to_lds_b128(
        (gv4i_p)(W + (size_t)(n0 + rA0) * K + k0 + kA0),
        (lv4i_p)(&Ws[buf][rA0 * LDSK + kA0]), 0, 0);
    __builtin_amdgcn_global_load_async_to_lds_b128(
        (gv4i_p)(W + (size_t)(n0 + rA1) * K + k0 + kA1),
        (lv4i_p)(&Ws[buf][rA1 * LDSK + kA1]), 0, 0);
  };

  issue_tile(0, 0);
  int buf = 0;
  for (int k0 = 0; k0 < K; k0 += BK, buf ^= 1) {
    __builtin_amdgcn_s_wait_asynccnt(0);  // this wave's DMA into buf done
    __syncthreads();                      // all waves' DMA done; prev consumers done
    if (k0 + BK < K) issue_tile(buf ^ 1, k0 + BK);  // DMA next tile during WMMAs
    mma_step(Xs[buf], Ws[buf]);
  }
#else
  // ---- fallback: register-pipelined staging (single LDS buffer) ----
  __shared__ __align__(16) u16 Xs1[BM * LDSK];
  __shared__ __align__(16) u16 Ws1[BN * LDSK];

  uint4 rx0, rx1, rw0, rw1;
  auto fetch_tile = [&](int k0) {
    rx0 = *reinterpret_cast<const uint4*>(X + (size_t)(m0 + rA0) * K + k0 + kA0);
    rx1 = *reinterpret_cast<const uint4*>(X + (size_t)(m0 + rA1) * K + k0 + kA1);
    rw0 = *reinterpret_cast<const uint4*>(W + (size_t)(n0 + rA0) * K + k0 + kA0);
    rw1 = *reinterpret_cast<const uint4*>(W + (size_t)(n0 + rA1) * K + k0 + kA1);
  };

  fetch_tile(0);
  for (int k0 = 0; k0 < K; k0 += BK) {
    __syncthreads();  // previous tile fully consumed
    *reinterpret_cast<uint4*>(&Xs1[rA0 * LDSK + kA0]) = rx0;
    *reinterpret_cast<uint4*>(&Xs1[rA1 * LDSK + kA1]) = rx1;
    *reinterpret_cast<uint4*>(&Ws1[rA0 * LDSK + kA0]) = rw0;
    *reinterpret_cast<uint4*>(&Ws1[rA1 * LDSK + kA1]) = rw1;
    __syncthreads();
    if (k0 + BK < K) {
      fetch_tile(k0 + BK);  // next slab in flight while WMMAs run
      __builtin_prefetch(X + (size_t)(m0 + (tid >> 1)) * K + k0 + 2 * BK, 0, 0);
      __builtin_prefetch(W + (size_t)(n0 + (tid >> 1)) * K + k0 + 2 * BK, 0, 0);
    }
    mma_step(Xs1, Ws1);
  }
#endif

  float scale = 1.0f;
  if (inv_alpha) scale = 1.0f / alpha_ptr[0];
  float w0 = 0.f, w1 = 0.f;
  if (EPI == 2) {
    float al = alpha_ptr[0];
    w0 = lam0[0] / al;
    w1 = lam1[0] / al;
  }

  // C/D layout: VGPR v, lane l -> m = v + (l>>4)*8, n = l&15
  const int mb = (lane >> 4) * 8;
  const int nn = lane & 15;
#pragma unroll
  for (int mt = 0; mt < 4; ++mt) {
#pragma unroll
    for (int nt = 0; nt < 2; ++nt) {
      int mbase = m0 + wy * 64 + mt * 16 + mb;
      int nidx  = n0 + wx * 32 + nt * 16 + nn;
#pragma unroll
      for (int v = 0; v < 8; ++v) {
        size_t idx = (size_t)(mbase + v) * N + nidx;
        float r = acc[mt][nt][v] * scale;
        if (EPI == 0) {
          ((float*)outp)[idx] = r;
        } else if (EPI == 1) {
          ((u16*)outp)[idx] = f32_to_bf16(r);
        } else {
          float h = prox_l1l1(r + xv[idx], w0, w1, a1[idx]);
          ((u16*)outp)[idx] = f32_to_bf16(h);
        }
      }
    }
  }
}

// ---------------------------------------------------------------------------
// Small helper kernels
// ---------------------------------------------------------------------------
__global__ void convert_f32_to_bf16(const float* __restrict__ in, u16* __restrict__ out, int n) {
  int i = blockIdx.x * blockDim.x + threadIdx.x;
  if (i < n) out[i] = f32_to_bf16(in[i]);
}

// in f32 [R,C] -> out bf16 [C,R]; R,C multiples of 32
__global__ void transpose_f32_to_bf16(const float* __restrict__ in, u16* __restrict__ out,
                                      int R, int C) {
  __shared__ float tile[32][33];
  int bx = blockIdx.x * 32;  // col base
  int by = blockIdx.y * 32;  // row base
  int tx = threadIdx.x;      // 0..31
  int ty = threadIdx.y;      // 0..7
#pragma unroll
  for (int i = 0; i < 32; i += 8)
    tile[ty + i][tx] = in[(size_t)(by + ty + i) * C + bx + tx];
  __syncthreads();
#pragma unroll
  for (int i = 0; i < 32; i += 8)
    out[(size_t)(bx + ty + i) * R + by + tx] = f32_to_bf16(tile[tx][ty + i]);
}

__global__ void build_S_temp(const float* __restrict__ tempf, u16* __restrict__ tempbf,
                             u16* __restrict__ Sbf, int nh) {
  int i = blockIdx.x * blockDim.x + threadIdx.x;
  if (i >= nh * nh) return;
  float t = tempf[i];
  tempbf[i] = f32_to_bf16(t);
  int r = i / nh, c = i - r * nh;
  Sbf[i] = f32_to_bf16((r == c ? 1.0f : 0.0f) - t);
}

__global__ void build_W1(const float* __restrict__ Gf, const float* __restrict__ tempG,
                         u16* __restrict__ W1bf, int n) {
  int i = blockIdx.x * blockDim.x + threadIdx.x;
  if (i < n) W1bf[i] = f32_to_bf16(Gf[i] - tempG[i]);
}

// ---------------------------------------------------------------------------
// Host orchestration
// ---------------------------------------------------------------------------
static void gemm_launch(int epi, const u16* X, const u16* W, void* out, int M, int N, int K,
                        const float* alpha, int inv, const float* xv, const float* a1,
                        const float* l0, const float* l1, hipStream_t s) {
  dim3 g(N / BN, M / BM), b(256);
  if (epi == 0)
    gemm_nt<0><<<g, b, 0, s>>>(X, W, out, M, N, K, alpha, inv, xv, a1, l0, l1);
  else if (epi == 1)
    gemm_nt<1><<<g, b, 0, s>>>(X, W, out, M, N, K, alpha, inv, xv, a1, l0, l1);
  else
    gemm_nt<2><<<g, b, 0, s>>>(X, W, out, M, N, K, alpha, inv, xv, a1, l0, l1);
}

extern "C" void kernel_launch(void* const* d_in, const int* in_sizes, int n_in,
                              void* d_out, int out_size, void* d_ws, size_t ws_size,
                              hipStream_t stream) {
  constexpr int T = 16, B = 512, NF = 1024, NI = 256, NH = 2048;
  // K_STEPS = 5 (harness constant; host loop count cannot depend on device data
  // under graph capture).
  (void)in_sizes; (void)n_in; (void)out_size; (void)ws_size;

  // d_in[0] = pre_input : provably unused (concat(...)[1:] drops it)
  const float* raw   = (const float*)d_in[1];  // [T,B,NF]
  const float* A     = (const float*)d_in[2];  // [NI,NF]
  const float* D     = (const float*)d_in[3];  // [NF,NH]
  const float* G     = (const float*)d_in[4];  // [NH,NH]
  const float* h0    = (const float*)d_in[5];  // [B,NH]
  const float* alpha = (const float*)d_in[6];
  const float* lam0  = (const float*)d_in[7];
  const float* lam1  = (const float*)d_in[8];

  char* base = (char*)d_ws;
  size_t off = 0;
  auto alloc = [&](size_t bytes) -> void* {
    void* p = base + off;
    off = (off + bytes + 255) & ~(size_t)255;
    return p;
  };

  // bf16 operand buffers (all row-major, K-contiguous for NT GEMM)
  u16* Abf    = (u16*)alloc((size_t)NI * NF * 2);        // A        [NI,NF]
  u16* Atbf   = (u16*)alloc((size_t)NF * NI * 2);        // A^T      [NF,NI]
  u16* Dbf    = (u16*)alloc((size_t)NF * NH * 2);        // D        [NF,NH]
  u16* Dtbf   = (u16*)alloc((size_t)NH * NF * 2);        // D^T      [NH,NF]
  u16* Gtbf   = (u16*)alloc((size_t)NH * NH * 2);        // G^T
  u16* rawbf  = (u16*)alloc((size_t)T * B * NF * 2);     // raw input
  u16* AtAbf  = (u16*)alloc((size_t)NF * NF * 2);        // A^T A
  u16* ADtbf  = (u16*)alloc((size_t)NH * NF * 2);        // (AtA D)^T
  u16* tempbf = (u16*)alloc((size_t)NH * NH * 2);
  u16* Sbf    = (u16*)alloc((size_t)NH * NH * 2);        // I - temp
  u16* W1bf   = (u16*)alloc((size_t)NH * NH * 2);        // G - temp G
  u16* Vbf    = (u16*)alloc((size_t)NH * NI * 2);        // V = D^T A^T / a
  u16* xallbf = (u16*)alloc((size_t)T * B * NI * 2);     // raw @ A^T
  u16* hs     = (u16*)alloc((size_t)(T + 1) * B * NH * 2);  // h history (slot0=h0)
  u16* hA     = (u16*)alloc((size_t)B * NH * 2);
  u16* hB     = (u16*)alloc((size_t)B * NH * 2);
  // f32 scratch
  float* ADf32   = (float*)alloc((size_t)NF * NH * 4);   // AtA @ D
  float* tempf32 = (float*)alloc((size_t)NH * NH * 4);   // temp, then reused for temp@G
  float* xv      = (float*)alloc((size_t)T * B * NH * 4);  // all xv_t, hoisted
  float* a1f     = (float*)alloc((size_t)B * NH * 4);

  auto cvt = [&](const float* in, u16* out, int n) {
    convert_f32_to_bf16<<<dim3((n + 255) / 256), dim3(256), 0, stream>>>(in, out, n);
  };
  auto tr = [&](const float* in, u16* out, int R, int C) {
    transpose_f32_to_bf16<<<dim3(C / 32, R / 32), dim3(32, 8), 0, stream>>>(in, out, R, C);
  };

  // ---- Phase 0: convert / transpose inputs to bf16 ----
  cvt(A, Abf, NI * NF);
  tr(A, Atbf, NI, NF);
  cvt(D, Dbf, NF * NH);
  tr(D, Dtbf, NF, NH);
  tr(G, Gtbf, NH, NH);
  cvt(raw, rawbf, T * B * NF);
  cvt(h0, hs, B * NH);  // slot 0 = h_0

  // ---- Phase 1: precompute (all hoisted out of the time loop) ----
  // AtA = At @ At^T                     [NF,NF]
  gemm_launch(1, Atbf, Atbf, AtAbf, NF, NF, NI, alpha, 0, 0, 0, 0, 0, stream);
  // AD = AtA @ D  (op2 = D^T)          [NF,NH] f32
  gemm_launch(0, AtAbf, Dtbf, ADf32, NF, NH, NF, alpha, 0, 0, 0, 0, 0, stream);
  tr(ADf32, ADtbf, NF, NH);
  // temp = (D^T @ AD)/a  (symmetric)   [NH,NH] f32
  gemm_launch(0, Dtbf, ADtbf, tempf32, NH, NH, NF, alpha, 1, 0, 0, 0, 0, stream);
  build_S_temp<<<dim3((NH * NH + 255) / 256), dim3(256), 0, stream>>>(tempf32, tempbf, Sbf, NH);
  // tempG = temp @ G  (op2 = G^T); reuse tempf32 for output
  gemm_launch(0, tempbf, Gtbf, tempf32, NH, NH, NH, alpha, 0, 0, 0, 0, 0, stream);
  build_W1<<<dim3((NH * NH + 255) / 256), dim3(256), 0, stream>>>(G, tempf32, W1bf, NH * NH);
  // V = (D^T @ A^T)/a  (op2 = A)       [NH,NI]
  gemm_launch(1, Dtbf, Abf, Vbf, NH, NI, NF, alpha, 1, 0, 0, 0, 0, stream);
  // x_all = raw @ A^T  (op2 = A)       [T*B,NI]
  gemm_launch(1, rawbf, Abf, xallbf, T * B, NI, NF, alpha, 0, 0, 0, 0, 0, stream);
  // xv_all = x_all @ V^T  (op2 = V)    [T*B,NH] f32 — h-independent, hoisted
  gemm_launch(0, xallbf, Vbf, xv, T * B, NH, NI, alpha, 0, 0, 0, 0, 0, stream);

  // ---- Phase 2: the serial recurrence (T steps x K=5 prox-GEMMs) ----
  for (int t = 0; t < T; ++t) {
    const u16* hprev = hs + (size_t)t * B * NH;
    u16* hout        = hs + (size_t)(t + 1) * B * NH;
    const float* xvt = xv + (size_t)t * B * NH;
    // a1 = h_prev @ G  (op2 = G^T)
    gemm_launch(0, hprev, Gtbf, a1f, B, NH, NH, alpha, 0, 0, 0, 0, 0, stream);
    // h = prox(h_prev @ W1^T + xv)  (op2 = W1)
    gemm_launch(2, hprev, W1bf, hA, B, NH, NH, alpha, 0, xvt, a1f, lam0, lam1, stream);
    // K-1 = 4 iterations: h = prox(xv + h @ S^T)  (op2 = S), ping-pong buffers
    gemm_launch(2, hA, Sbf, hB, B, NH, NH, alpha, 0, xvt, a1f, lam0, lam1, stream);
    gemm_launch(2, hB, Sbf, hA, B, NH, NH, alpha, 0, xvt, a1f, lam0, lam1, stream);
    gemm_launch(2, hA, Sbf, hB, B, NH, NH, alpha, 0, xvt, a1f, lam0, lam1, stream);
    gemm_launch(2, hB, Sbf, hout, B, NH, NH, alpha, 0, xvt, a1f, lam0, lam1, stream);
  }

  // ---- Phase 3: z_hat = hs @ D^T  (op2 = D) -> d_out f32 [T*B,NF] ----
  gemm_launch(0, hs + (size_t)B * NH, Dbf, d_out, T * B, NF, NH, alpha, 0, 0, 0, 0, 0, stream);
}
